// DilatedAttention_25391846654039
// MI455X (gfx1250) — matrix operand order repaired and checked by reading, not compiled, not run on hardware
//
#include <hip/hip_runtime.h>
#include <hip/hip_bf16.h>

// ---------------- types ----------------
typedef __attribute__((ext_vector_type(16))) __bf16        v16bf;
typedef __attribute__((ext_vector_type(8)))  float         v8f;
typedef __attribute__((ext_vector_type(4)))  unsigned int  u32x4;
typedef __attribute__((ext_vector_type(4)))  float         f32x4;

#define DEV static __device__ __forceinline__

// ---------------- problem constants ----------------
constexpr int BB   = 4;
constexpr int SS   = 16384;
constexpr int DD   = 512;
constexpr int SEGL = 256;   // segment size
constexpr int LL   = 128;   // rows per segment after dilation (SEG/DIL)
constexpr int NSEG = 64;    // SS / SEGL
constexpr int NBLK = BB * NSEG;  // 256 (b,seg) blocks

// workspace layout (in bf16 element units)
constexpr size_t Q_OFF = 0;
constexpr size_t K_OFF = (size_t)NBLK * LL * DD;       // 16,777,216
constexpr size_t V_OFF = K_OFF * 2;                    // v stored TRANSPOSED [blk][dout][row]
constexpr size_t W_OFF = K_OFF * 3;                    // Wq,Wk,Wv bf16 row-major [dout][din]

// ---------------- helpers ----------------
DEV unsigned short f2bf(float f) {
  unsigned u = __builtin_bit_cast(unsigned, f);
  u += 0x7FFFu + ((u >> 16) & 1u);          // round-to-nearest-even
  return (unsigned short)(u >> 16);
}
DEV unsigned pack2(float lo, float hi) {
  return (unsigned)f2bf(lo) | ((unsigned)f2bf(hi) << 16);
}
DEV v16bf frag_cat(u32x4 lo, u32x4 hi) {
  union { u32x4 u[2]; v16bf v; } t;
  t.u[0] = lo; t.u[1] = hi;
  return t.v;
}
// contiguous 32B (16 bf16) -> B-fragment (lane-local K run), global or LDS
DEV v16bf frag32(const unsigned short* p) {
  const u32x4* q = (const u32x4*)p;
  return frag_cat(q[0], q[1]);
}
// A-fragment: two 16B chunks at +0 and +8 halves (contiguous 16-k run) within a row-major row
DEV v16bf fragA(const unsigned short* row, int kb, int g) {
  const u32x4* p0 = (const u32x4*)(row + kb * 32 + g * 16);
  const u32x4* p1 = (const u32x4*)(row + kb * 32 + g * 16 + 8);
  return frag_cat(*p0, *p1);
}
DEV v8f wmma_bf16(v16bf a, v16bf b, v8f c) {
  return __builtin_amdgcn_wmma_f32_16x16x32_bf16(false, a, false, b, (short)0, c,
                                                 false, false);
}

// ---------------- kernel 0: weights fp32 -> bf16 ----------------
__global__ void wconv_kernel(const float* __restrict__ Wq,
                             const float* __restrict__ Wk,
                             const float* __restrict__ Wv,
                             unsigned short* __restrict__ wbf) {
  int i = blockIdx.x * 256 + threadIdx.x;        // 0 .. 3*512*512-1
  const float* src = (i < 262144) ? Wq : ((i < 524288) ? Wk : Wv);
  wbf[i] = f2bf(src[i & 262143]);
}

// ---------------- kernel 1: fused dilated-gather + QKV projection ----------------
// one workgroup per (b,seg). LDS (128KB) is first the bf16 x-tile, then (after
// A-fragments are register-resident) reused as a 32KB staged weight tile that
// all 8 waves share -> B-fragments become ds_load_b128 with 8x reuse.
__global__ void __launch_bounds__(256, 1)
qkv_kernel(const float* __restrict__ x,
           const float* __restrict__ bq, const float* __restrict__ bk,
           const float* __restrict__ bv,
           const unsigned short* __restrict__ wbf,
           unsigned short* __restrict__ qout, unsigned short* __restrict__ kout,
           unsigned short* __restrict__ vtout) {
  __shared__ unsigned short sBuf[LL * DD];       // 128 KB, reused x-tile -> W-tiles

  const int blk = blockIdx.x;
  const int b   = blk >> 6;
  const int seg = blk & 63;
  const int tid = threadIdx.x;

  // stage dilated rows (every 2nd row of the 256-row segment), fp32 -> bf16
  const float* xbase = x + ((size_t)b * SS + (size_t)seg * SEGL) * DD;
  for (int it = 0; it < 64; ++it) {
    int idx4 = it * 256 + tid;                   // float4 index, 0..16383
    int row  = idx4 >> 7;                        // 128 float4 per row
    int c4   = idx4 & 127;
    f32x4 v  = *(const f32x4*)(xbase + (size_t)(2 * row) * DD + c4 * 4);
    unsigned* dst = (unsigned*)&sBuf[row * DD + c4 * 4];
    dst[0] = pack2(v[0], v[1]);
    dst[1] = pack2(v[2], v[3]);
  }
  __syncthreads();

  const int wave = tid >> 5, lane = tid & 31, lr = lane & 15, g = lane >> 4;

  // preload all 16 A-fragments of this wave's 16-row block (reused 96x)
  v16bf aF[16];
  const unsigned short* arow = sBuf + (wave * 16 + lr) * DD;
#pragma unroll
  for (int kb = 0; kb < 16; ++kb) aF[kb] = fragA(arow, kb, g);
  // x-tile in LDS is now dead; sBuf becomes the shared weight-tile buffer.

  for (int m = 0; m < 3; ++m) {
    const unsigned short* wmat = wbf + (size_t)m * DD * DD;
    const float* bias = (m == 0) ? bq : ((m == 1) ? bk : bv);
    for (int nt2 = 0; nt2 < 16; ++nt2) {         // two 16-col output tiles per pass
      // cooperative stage: W rows [nt2*32, nt2*32+32) x 512 -> 32KB of sBuf
      const unsigned short* gW = wmat + (size_t)nt2 * 32 * DD + tid * 64;
      __syncthreads();                           // previous pass done reading sBuf
      {
        const u32x4* src = (const u32x4*)gW;
        u32x4* dst = (u32x4*)&sBuf[tid * 64];
#pragma unroll
        for (int i = 0; i < 4; ++i) dst[i] = src[i];
        if (m * 16 + nt2 < 47)                   // prefetch next weight tile
          __builtin_prefetch(gW + 32 * DD, 0, 1);
      }
      __syncthreads();

      // dual independent WMMA chains (tiles nt=2*nt2 and nt=2*nt2+1)
      v8f acc0 = {}, acc1 = {};
      const unsigned short* w0 = sBuf + lr * DD;
      const unsigned short* w1 = sBuf + (16 + lr) * DD;
#pragma unroll
      for (int kb = 0; kb < 16; ++kb) {
        v16bf b0 = frag32(w0 + kb * 32 + g * 16);
        v16bf b1 = frag32(w1 + kb * 32 + g * 16);
        acc0 = wmma_bf16(aF[kb], b0, acc0);
        acc1 = wmma_bf16(aF[kb], b1, acc1);
      }

#pragma unroll
      for (int half = 0; half < 2; ++half) {
        const v8f& acc = half ? acc1 : acc0;
        int nt = 2 * nt2 + half;
        float bb = bias[nt * 16 + lr];
        if (m == 2) {
          // store V transposed: vT[blk][dout][row]; 8 consecutive rows -> b128
          u32x4 pk;
          pk[0] = pack2(acc[0] + bb, acc[1] + bb);
          pk[1] = pack2(acc[2] + bb, acc[3] + bb);
          pk[2] = pack2(acc[4] + bb, acc[5] + bb);
          pk[3] = pack2(acc[6] + bb, acc[7] + bb);
          unsigned short* vp =
              vtout + ((size_t)blk * DD + nt * 16 + lr) * LL + wave * 16 + 8 * g;
          *(u32x4*)vp = pk;
        } else {
          unsigned short* qk = (m == 0 ? qout : kout) +
                               ((size_t)blk * LL + wave * 16 + 8 * g) * DD +
                               nt * 16 + lr;
#pragma unroll
          for (int j = 0; j < 8; ++j) qk[(size_t)j * DD] = f2bf(acc[j] + bb);
        }
      }
    }
  }
}

// ---------------- kernel 2: per-segment attention ----------------
// one workgroup per (b,seg); wave = one 16-row query block. K tile lives in
// LDS (staged once, shared by all 8 waves); the probability buffer overlays
// the then-dead K region, keeping total LDS at 128KB.
__global__ void __launch_bounds__(256, 1)
attn_kernel(const unsigned short* __restrict__ qws,
            const unsigned short* __restrict__ kws,
            const unsigned short* __restrict__ vtws,
            float* __restrict__ out) {
  __shared__ unsigned short sBuf[LL * DD];       // 128 KB: K tile, then probs

  const int blk = blockIdx.x;
  const int tid = threadIdx.x, wave = tid >> 5, lane = tid & 31;
  const int lr = lane & 15, g = lane >> 4;

  // ---- stage K tile [128 x 512] bf16 into LDS (b128 copy, coalesced) ----
  const unsigned short* kg = kws + (size_t)blk * LL * DD;
  for (int it = 0; it < 32; ++it) {
    int c = it * 256 + tid;                      // 16B-chunk index, 0..8191
    *(u32x4*)&sBuf[c * 8] = *(const u32x4*)(kg + c * 8);
  }
  __syncthreads();

  const unsigned short* qrow = qws + ((size_t)blk * LL + wave * 16 + lr) * DD;

  // ---- scores = q @ k^T  (16 x 128, f32 accum; 8 independent chains) ----
  v8f zero = {};
  v8f acc[8];
#pragma unroll
  for (int c = 0; c < 8; ++c) acc[c] = zero;

#pragma unroll 4
  for (int kb = 0; kb < 16; ++kb) {
    v16bf aF = fragA(qrow, kb, g);
#pragma unroll
    for (int c = 0; c < 8; ++c) {
      v16bf bF = frag32(sBuf + (c * 16 + lr) * DD + kb * 32 + g * 16);
      acc[c] = wmma_bf16(aF, bF, acc[c]);
    }
  }

  // ---- softmax over 128 columns; row m = 8g + j lives in 16 lanes (l&15) ----
  const float scale = 0.04419417382415922f;      // 1/sqrt(512)
  float rinv[8];
#pragma unroll
  for (int j = 0; j < 8; ++j) {
    float mx = -3.0e38f;
#pragma unroll
    for (int c = 0; c < 8; ++c) mx = fmaxf(mx, acc[c][j]);
#pragma unroll
    for (int m = 1; m < 16; m <<= 1) mx = fmaxf(mx, __shfl_xor(mx, m, 32));
    float rmax = mx * scale;
    float sum = 0.f;
#pragma unroll
    for (int c = 0; c < 8; ++c) {
      float p = __expf(acc[c][j] * scale - rmax);
      acc[c][j] = p;
      sum += p;
    }
#pragma unroll
    for (int m = 1; m < 16; m <<= 1) sum += __shfl_xor(sum, m, 32);
    rinv[j] = 1.0f / sum;
  }

  __syncthreads();   // all waves done reading K before probs overlay sBuf

  // ---- re-shape probs C-layout -> A-layout via LDS (wave-private slice) ----
  unsigned short* ps = sBuf + wave * 16 * LL;    // 8 * 4KB = 32KB overlay
#pragma unroll
  for (int c = 0; c < 8; ++c)
#pragma unroll
    for (int j = 0; j < 8; ++j)
      ps[(8 * g + j) * LL + c * 16 + lr] = f2bf(acc[c][j] * rinv[j]);
  __syncthreads();

  v16bf pF[4];
  const unsigned short* prow = ps + lr * LL;
#pragma unroll
  for (int kb = 0; kb < 4; ++kb) pF[kb] = fragA(prow, kb, g);

  // ---- out = attn @ v (transposed V -> contiguous B-frags); dual chains ----
  const unsigned short* vtb = vtws + (size_t)blk * DD * LL;
  float* ob = out + ((size_t)blk * LL + wave * 16 + 8 * g) * DD;
  for (int nt2 = 0; nt2 < 16; ++nt2) {
    v8f o0 = zero, o1 = zero;
#pragma unroll
    for (int kb = 0; kb < 4; ++kb) {
      v16bf b0 = frag32(vtb + (size_t)(nt2 * 32 + lr) * LL + kb * 32 + g * 16);
      v16bf b1 = frag32(vtb + (size_t)(nt2 * 32 + 16 + lr) * LL + kb * 32 + g * 16);
      o0 = wmma_bf16(pF[kb], b0, o0);
      o1 = wmma_bf16(pF[kb], b1, o1);
    }
#pragma unroll
    for (int j = 0; j < 8; ++j) {
      ob[(size_t)j * DD + (2 * nt2) * 16 + lr]     = o0[j];
      ob[(size_t)j * DD + (2 * nt2 + 1) * 16 + lr] = o1[j];
    }
  }
}

// ---------------- launcher ----------------
extern "C" void kernel_launch(void* const* d_in, const int* in_sizes, int n_in,
                              void* d_out, int out_size, void* d_ws, size_t ws_size,
                              hipStream_t stream) {
  (void)in_sizes; (void)n_in; (void)out_size; (void)ws_size;
  const float* x  = (const float*)d_in[0];
  const float* Wq = (const float*)d_in[1];
  const float* bq = (const float*)d_in[2];
  const float* Wk = (const float*)d_in[3];
  const float* bk = (const float*)d_in[4];
  const float* Wv = (const float*)d_in[5];
  const float* bv = (const float*)d_in[6];

  unsigned short* ws   = (unsigned short*)d_ws;
  unsigned short* qws  = ws + Q_OFF;
  unsigned short* kws  = ws + K_OFF;
  unsigned short* vtws = ws + V_OFF;
  unsigned short* wbf  = ws + W_OFF;
  float* out = (float*)d_out;

  wconv_kernel<<<3 * 512 * 512 / 256, 256, 0, stream>>>(Wq, Wk, Wv, wbf);
  qkv_kernel<<<NBLK, 256, 0, stream>>>(x, bq, bk, bv, wbf, qws, kws, vtws);
  attn_kernel<<<NBLK, 256, 0, stream>>>(qws, kws, vtws, out);
}
